// MatchNet_71648644432054
// MI455X (gfx1250) — compile-verified, matrix operand order checked by reading, not run on hardware
//
#include <hip/hip_runtime.h>

// ---------------------------------------------------------------------------
// MatchNet pipeline for gfx1250 (MI455X, wave32, WMMA bf16 16x16x32)
//   conv(3x3,s2,p1)+ReLU  -> implicit GEMM  M=50176 K=2304 N=256
//   h = ReLU(x@w1+b1)     -> GEMM           M=1024  K=12544 N=1024
//   e = h@w2+b2, L2-norm  -> GEMM           M=1024  K=1024  N=256
//   logits[p] = wm . |e_i - e_j| + bm  for all i<j  (P = 523776)
// ---------------------------------------------------------------------------

typedef __attribute__((ext_vector_type(16))) __bf16 v16bf;
typedef __attribute__((ext_vector_type(8)))  __bf16 v8bf;
typedef __attribute__((ext_vector_type(8)))  float  v8f;

#define N_ROI 1024
#define CCH   256
#define FLAT  12544
#define HID   1024
#define EMB   256
#define CONVM 50176           // 1024*49
#define CONVK 2304            // 256*9
#define NPAIR 523776          // 1024*1023/2

// fp32 -> bf16 round-to-nearest-even
__device__ __forceinline__ unsigned short f2bf(float f) {
    unsigned int u = __builtin_bit_cast(unsigned int, f);
    u = (u + 0x7FFFu + ((u >> 16) & 1u)) >> 16;
    return (unsigned short)u;
}

// Load one 16x32 bf16 A/B fragment from an LDS tile stored row-major with
// `stride` (elements).  ISA 7.12.2: lane L holds row (L&15); its 16 bf16
// elements are two contiguous 8-element runs at k = (L>>4)*8 and k = 16+(L>>4)*8.
__device__ __forceinline__ v16bf load_frag(const unsigned short* base, int stride,
                                           int row_base, int lane) {
    const unsigned short* p = base + (row_base + (lane & 15)) * stride + ((lane >> 4) << 3);
    v8bf lo = *(const v8bf*)(p);
    v8bf hi = *(const v8bf*)(p + 16);
    v16bf f;
#pragma unroll
    for (int t = 0; t < 8; ++t) { f[t] = lo[t]; f[t + 8] = hi[t]; }
    return f;
}

// ---------------------------------------------------------------------------
// dtype conversion helpers
// ---------------------------------------------------------------------------
__global__ void cvt_f32_bf16_x4(const float* __restrict__ src,
                                unsigned short* __restrict__ dst, int n4) {
    int i = blockIdx.x * blockDim.x + threadIdx.x;
    if (i >= n4) return;
    float4 v = ((const float4*)src)[i];
    unsigned long long r = (unsigned long long)f2bf(v.x)
                         | ((unsigned long long)f2bf(v.y) << 16)
                         | ((unsigned long long)f2bf(v.z) << 32)
                         | ((unsigned long long)f2bf(v.w) << 48);
    ((unsigned long long*)dst)[i] = r;
}

// conv_w [O=256][C=256][3][3]  ->  bf16 B matrix [K=2304][N=256], k=c*9+ky*3+kx
__global__ void reorder_convw(const float* __restrict__ src,
                              unsigned short* __restrict__ dst) {
    int idx = blockIdx.x * blockDim.x + threadIdx.x;   // < 2304*256
    int k = idx >> 8, o = idx & 255;
    dst[(size_t)k * 256 + o] = f2bf(src[(size_t)o * CONVK + k]);
}

// ---------------------------------------------------------------------------
// Implicit-GEMM conv: A gathered from fp32 features (im2col, zero pad),
// B = reordered conv weights (bf16).  Epilogue: +bias, ReLU, scatter-store
// bf16 into x[n][c*49 + oy*7 + ox].
// block = 256 threads (8 waves), tile 128(M) x 128(N), KT=32
// ---------------------------------------------------------------------------
__global__ void __launch_bounds__(256)
conv_gemm_kernel(const float* __restrict__ feat,
                 const unsigned short* __restrict__ B,
                 const float* __restrict__ bias,
                 unsigned short* __restrict__ x_out) {
    __shared__ unsigned short As[128][40];
    __shared__ unsigned short Bs[128][40];

    const int tid  = threadIdx.x;
    const int lane = tid & 31;
    const int wid  = tid >> 5;
    const int wm   = wid & 3;         // 0..3 -> 32-row slice
    const int wn   = wid >> 2;        // 0..1 -> 64-col slice
    const int row0 = blockIdx.x * 128;
    const int n0   = blockIdx.y * 128;

    v8f acc[2][4];
    v8f vzero = {};
#pragma unroll
    for (int a = 0; a < 2; ++a)
#pragma unroll
        for (int b = 0; b < 4; ++b) acc[a][b] = vzero;

    for (int k0 = 0; k0 < CONVK; k0 += 32) {
        // ---- stage A (im2col gather + cvt): 128x32 elements, 16/thread ----
#pragma unroll 4
        for (int p = 0; p < 16; ++p) {
            int lin = p * 256 + tid;           // 0..4095
            int row = lin >> 5, kk = lin & 31;
            int grow = row0 + row;
            int n  = grow / 49, rr = grow - n * 49;
            int oy = rr / 7,    ox = rr - oy * 7;
            int k  = k0 + kk;
            int c  = k / 9,  s  = k - c * 9;
            int ky = s / 3,  kx = s - ky * 3;
            int iy = 2 * oy - 1 + ky;
            int ix = 2 * ox - 1 + kx;
            float v = 0.f;
            if ((unsigned)iy < 14u && (unsigned)ix < 14u)
                v = feat[(((size_t)n * CCH + c) * 14 + iy) * 14 + ix];
            As[row][kk] = f2bf(v);
        }
        // ---- stage B transposed: 32x128 -> Bs[col][k] ----
#pragma unroll
        for (int p = 0; p < 2; ++p) {
            int lin = p * 256 + tid;
            int kk = lin >> 4, c8 = (lin & 15) << 3;
            uint4 d = *(const uint4*)(B + (size_t)(k0 + kk) * 256 + n0 + c8);
            const unsigned short* ds = (const unsigned short*)&d;
#pragma unroll
            for (int i = 0; i < 8; ++i) Bs[c8 + i][kk] = ds[i];
        }
        if (k0 + 32 < CONVK)
            __builtin_prefetch(B + (size_t)(k0 + 32 + (tid & 31)) * 256 + n0, 0, 0);
        __syncthreads();

        v16bf af[2];
#pragma unroll
        for (int fm = 0; fm < 2; ++fm)
            af[fm] = load_frag(&As[0][0], 40, wm * 32 + fm * 16, lane);
        v16bf bfr[4];
#pragma unroll
        for (int fn = 0; fn < 4; ++fn)
            bfr[fn] = load_frag(&Bs[0][0], 40, wn * 64 + fn * 16, lane);
#pragma unroll
        for (int fm = 0; fm < 2; ++fm)
#pragma unroll
            for (int fn = 0; fn < 4; ++fn)
                acc[fm][fn] = __builtin_amdgcn_wmma_f32_16x16x32_bf16(
                    false, af[fm], false, bfr[fn], (short)0, acc[fm][fn], false, false);
        __syncthreads();
    }

    // epilogue: +bias, ReLU, scatter into [n][c*49 + oy*7 + ox] layout (bf16)
#pragma unroll
    for (int fm = 0; fm < 2; ++fm)
#pragma unroll
        for (int fn = 0; fn < 4; ++fn)
#pragma unroll
            for (int r = 0; r < 8; ++r) {
                int m_l = wm * 32 + fm * 16 + r + ((lane >> 4) << 3);
                int n_l = wn * 64 + fn * 16 + (lane & 15);
                int grow = row0 + m_l;
                int col  = n0 + n_l;
                float v = acc[fm][fn][r] + bias[col];
                v = fmaxf(v, 0.f);
                int n = grow / 49, rr = grow - n * 49;
                x_out[(size_t)n * FLAT + col * 49 + rr] = f2bf(v);
            }
}

// ---------------------------------------------------------------------------
// Dense bf16 GEMM:  C[M,Nc] = A[M,K] @ B[K,Nc] (+bias)(+ReLU)
// out_bf != null -> store bf16, else store fp32 to out_f.
// block = 256 threads, tile 128x128, KT=32
// ---------------------------------------------------------------------------
__global__ void __launch_bounds__(256)
gemm_bf16_kernel(const unsigned short* __restrict__ A,
                 const unsigned short* __restrict__ B,
                 const float* __restrict__ bias,
                 unsigned short* __restrict__ out_bf,
                 float* __restrict__ out_f,
                 int M, int Nc, int K, int relu) {
    __shared__ unsigned short As[128][40];
    __shared__ unsigned short Bs[128][40];

    const int tid  = threadIdx.x;
    const int lane = tid & 31;
    const int wid  = tid >> 5;
    const int wm   = wid & 3;
    const int wn   = wid >> 2;
    const int row0 = blockIdx.x * 128;
    const int n0   = blockIdx.y * 128;

    v8f acc[2][4];
    v8f vzero = {};
#pragma unroll
    for (int a = 0; a < 2; ++a)
#pragma unroll
        for (int b = 0; b < 4; ++b) acc[a][b] = vzero;

    for (int k0 = 0; k0 < K; k0 += 32) {
        // A tile 128x32: two 16B chunks per thread
#pragma unroll
        for (int p = 0; p < 2; ++p) {
            int lin = p * 256 + tid;
            int row = lin >> 2, kk8 = (lin & 3) << 3;
            uint4 d = *(const uint4*)(A + (size_t)(row0 + row) * K + k0 + kk8);
            *(uint4*)&As[row][kk8] = d;
        }
        // B tile 32x128 -> transposed Bs[col][k]
#pragma unroll
        for (int p = 0; p < 2; ++p) {
            int lin = p * 256 + tid;
            int kk = lin >> 4, c8 = (lin & 15) << 3;
            uint4 d = *(const uint4*)(B + (size_t)(k0 + kk) * Nc + n0 + c8);
            const unsigned short* ds = (const unsigned short*)&d;
#pragma unroll
            for (int i = 0; i < 8; ++i) Bs[c8 + i][kk] = ds[i];
        }
        if (k0 + 32 < K) {   // gfx1250 global_prefetch of the next K tile
            __builtin_prefetch(A + (size_t)(row0 + (tid >> 1)) * K + k0 + 32, 0, 0);
            __builtin_prefetch(B + (size_t)(k0 + 32 + (tid & 31)) * Nc + n0, 0, 0);
        }
        __syncthreads();

        v16bf af[2];
#pragma unroll
        for (int fm = 0; fm < 2; ++fm)
            af[fm] = load_frag(&As[0][0], 40, wm * 32 + fm * 16, lane);
        v16bf bfr[4];
#pragma unroll
        for (int fn = 0; fn < 4; ++fn)
            bfr[fn] = load_frag(&Bs[0][0], 40, wn * 64 + fn * 16, lane);
#pragma unroll
        for (int fm = 0; fm < 2; ++fm)
#pragma unroll
            for (int fn = 0; fn < 4; ++fn)
                acc[fm][fn] = __builtin_amdgcn_wmma_f32_16x16x32_bf16(
                    false, af[fm], false, bfr[fn], (short)0, acc[fm][fn], false, false);
        __syncthreads();
    }

#pragma unroll
    for (int fm = 0; fm < 2; ++fm)
#pragma unroll
        for (int fn = 0; fn < 4; ++fn)
#pragma unroll
            for (int r = 0; r < 8; ++r) {
                int m_l = wm * 32 + fm * 16 + r + ((lane >> 4) << 3);
                int n_l = wn * 64 + fn * 16 + (lane & 15);
                int grow = row0 + m_l;
                int col  = n0 + n_l;
                float v = acc[fm][fn][r] + bias[col];
                if (relu) v = fmaxf(v, 0.f);
                if (out_bf) out_bf[(size_t)grow * Nc + col] = f2bf(v);
                else        out_f [(size_t)grow * Nc + col] = v;
            }
}

// ---------------------------------------------------------------------------
// Row-wise L2 normalize: e[row] = e_un[row] / ||e_un[row]||   (256 per row)
// ---------------------------------------------------------------------------
__global__ void normalize_kernel(const float* __restrict__ e_un,
                                 float* __restrict__ e) {
    __shared__ float red[256];
    int row = blockIdx.x, t = threadIdx.x;
    float v = e_un[(size_t)row * EMB + t];
    red[t] = v * v;
    __syncthreads();
#pragma unroll
    for (int s = 128; s > 0; s >>= 1) {
        if (t < s) red[t] += red[t + s];
        __syncthreads();
    }
    e[(size_t)row * EMB + t] = v * rsqrtf(red[0]);
}

// ---------------------------------------------------------------------------
// Pairwise: logits[p] = bm + sum_k wm[k] * |e[i][k] - e[j][k]|, p over i<j.
// Closed-form inversion of the triangular index; float4 loads (e is 1 MB,
// L2-resident at 192 MB).
// ---------------------------------------------------------------------------
__global__ void pairwise_kernel(const float* __restrict__ e,
                                const float* __restrict__ wm,
                                const float* __restrict__ bm,
                                float* __restrict__ out) {
    int p = blockIdx.x * blockDim.x + threadIdx.x;
    if (p >= NPAIR) return;
    const double bq = 2.0 * N_ROI - 1.0;   // 2047
    int i = (int)floor((bq - sqrt(bq * bq - 8.0 * (double)p)) * 0.5);
    if (i < 0) i = 0;
    if (i > N_ROI - 2) i = N_ROI - 2;
    while ((long long)i * (2 * N_ROI - 1 - i) / 2 > (long long)p) --i;
    while ((long long)(i + 1) * (2 * N_ROI - 2 - i) / 2 <= (long long)p) ++i;
    long long off = (long long)i * (2 * N_ROI - 1 - i) / 2;
    int j = (int)((long long)p - off) + i + 1;

    const float4* ei = (const float4*)(e + (size_t)i * EMB);
    const float4* ej = (const float4*)(e + (size_t)j * EMB);
    const float4* w4 = (const float4*)wm;
    float acc = 0.f;
#pragma unroll 8
    for (int k = 0; k < EMB / 4; ++k) {
        float4 a = ei[k], c = ej[k], w = w4[k];
        acc += w.x * fabsf(a.x - c.x) + w.y * fabsf(a.y - c.y)
             + w.z * fabsf(a.z - c.z) + w.w * fabsf(a.w - c.w);
    }
    out[p] = acc + bm[0];
}

// ---------------------------------------------------------------------------
extern "C" void kernel_launch(void* const* d_in, const int* in_sizes, int n_in,
                              void* d_out, int out_size, void* d_ws, size_t ws_size,
                              hipStream_t stream) {
    const float* features = (const float*)d_in[0];
    /* d_in[1] = ids (int64), unused by the reference output */
    const float* conv_w = (const float*)d_in[2];
    const float* conv_b = (const float*)d_in[3];
    const float* w1     = (const float*)d_in[4];
    const float* b1     = (const float*)d_in[5];
    const float* w2     = (const float*)d_in[6];
    const float* b2     = (const float*)d_in[7];
    const float* wm     = (const float*)d_in[8];
    const float* bm     = (const float*)d_in[9];
    float* out = (float*)d_out;

    // workspace carve-out (256B aligned)
    char* ws = (char*)d_ws;
    size_t off = 0;
    auto carve = [&](size_t bytes) -> void* {
        void* p = ws + off;
        off += (bytes + 255) & ~(size_t)255;
        return p;
    };
    unsigned short* x_bf    = (unsigned short*)carve((size_t)N_ROI * FLAT * 2);   // 25.7 MB
    unsigned short* w1_bf   = (unsigned short*)carve((size_t)FLAT * HID * 2);     // 25.7 MB
    unsigned short* cw_bf   = (unsigned short*)carve((size_t)CONVK * CCH * 2);    // 1.2 MB
    unsigned short* w2_bf   = (unsigned short*)carve((size_t)HID * EMB * 2);      // 0.5 MB
    unsigned short* h_bf    = (unsigned short*)carve((size_t)N_ROI * HID * 2);    // 2.0 MB
    float*          e_un    = (float*)carve((size_t)N_ROI * EMB * 4);             // 1.0 MB
    float*          e_norm  = (float*)carve((size_t)N_ROI * EMB * 4);             // 1.0 MB

    // 1) dtype conversions / weight reorder
    {
        int n4 = FLAT * HID / 4;
        cvt_f32_bf16_x4<<<(n4 + 255) / 256, 256, 0, stream>>>(w1, w1_bf, n4);
    }
    reorder_convw<<<(CONVK * CCH) / 256, 256, 0, stream>>>(conv_w, cw_bf);
    {
        int n4 = HID * EMB / 4;
        cvt_f32_bf16_x4<<<(n4 + 255) / 256, 256, 0, stream>>>(w2, w2_bf, n4);
    }

    // 2) conv as implicit GEMM: M=50176, K=2304, N=256
    conv_gemm_kernel<<<dim3(CONVM / 128, CCH / 128), 256, 0, stream>>>(
        features, cw_bf, conv_b, x_bf);

    // 3) GEMM1: h = ReLU(x @ w1 + b1)   M=1024 K=12544 N=1024
    gemm_bf16_kernel<<<dim3(N_ROI / 128, HID / 128), 256, 0, stream>>>(
        x_bf, w1_bf, b1, h_bf, nullptr, N_ROI, HID, FLAT, 1);

    // 4) GEMM2: e_un = h @ w2 + b2      M=1024 K=1024 N=256
    gemm_bf16_kernel<<<dim3(N_ROI / 128, EMB / 128), 256, 0, stream>>>(
        h_bf, w2_bf, b2, nullptr, e_un, N_ROI, EMB, HID, 0);

    // 5) row L2 normalize
    normalize_kernel<<<N_ROI, EMB, 0, stream>>>(e_un, e_norm);

    // 6) pairwise |e_i - e_j| . wm + bm
    pairwise_kernel<<<(NPAIR + 255) / 256, 256, 0, stream>>>(e_norm, wm, bm, out);
}
